// SelectiveSSM_82471962018274
// MI455X (gfx1250) — matrix operand order repaired and checked by reading, not compiled, not run on hardware
//
#include <hip/hip_runtime.h>
#include <hip/hip_bf16.h>
#include <math.h>

// ---------------------------------------------------------------------------
// Mamba selective-SSM block for MI455X (gfx1250, wave32, WMMA).
//   * All GEMM operands pre-converted to f16 once; GEMM staging is a raw
//     byte copy via gfx1250 async DMA (global_load_async_to_lds_b128,
//     ASYNCcnt) into double-buffered LDS, overlapping DMA with v_wmma.
//   * N-boundary guard is a template parameter: only the x_proj GEMM
//     (N=160) pays the clamp/zero cost; all others are straight-line.
//   * Windowed scan (K=8): shift-register ring, 1 exp per (t,d,n) element.
// ---------------------------------------------------------------------------

typedef __attribute__((ext_vector_type(16))) _Float16 v16h;
typedef __attribute__((ext_vector_type(8)))  _Float16 v8h;
typedef __attribute__((ext_vector_type(8)))  float    v8f;

#define D_MODEL 512
#define D_INNER 1024
#define STATE   64
#define DT_RANK 32
#define SEQ_L   1024
#define K_WIN   8
#define XDBL_N  (DT_RANK + 2 * STATE)   // 160

#define USE_ASYNC_COPY 1

// ------------------------------ WMMA GEMM ----------------------------------
// C[M,N] = A[M,K] (f16 row major, lda) * B[N,K]^T (f16 row major, ldb == K)
// Block tile 128x64, BK=32 (one WMMA K step / stage), 8 waves in 4(M)x2(N),
// each wave owns a 32x32 patch = 2x2 fragments.

#define BM 128
#define BN 64
#define BK 32
#define PADH 40                         // halves per LDS row (80 B, 16B mult)
#define PADB 80

enum { EPI_NONE = 0, EPI_SOFTPLUS_BIAS = 1, EPI_RESID = 2 };

__device__ __forceinline__ void async_copy_b128(uint32_t lds_addr,
                                                uint32_t byte_off,
                                                uint64_t base) {
#if USE_ASYNC_COPY
  asm volatile("global_load_async_to_lds_b128 %0, %1, %2"
               :: "v"(lds_addr), "v"(byte_off), "s"(base)
               : "memory");
#else
  *(__attribute__((address_space(3))) v8h*)(uintptr_t)lds_addr =
      *(const v8h*)((const char*)base + byte_off);
#endif
}

#if USE_ASYNC_COPY
#define ASYNC_WAIT(n) asm volatile("s_wait_asynccnt " #n ::: "memory")
#else
#define ASYNC_WAIT(n)
#endif

template <int EPI, bool W16, bool NGUARD>
__global__ __launch_bounds__(256)
void gemm_nt_wmma(const _Float16* __restrict__ A, int lda,
                  const _Float16* __restrict__ B,
                  float* __restrict__ C, int ldc,
                  _Float16* __restrict__ C16,
                  int M, int N, int K,
                  const float* __restrict__ bias,
                  const float* __restrict__ resid, int ldr) {
  __shared__ __align__(16) _Float16 sA[2][BM][PADH];
  __shared__ __align__(16) _Float16 sB[2][BN][PADH];

  const int tid  = threadIdx.x;
  const int lane = tid & 31;
  const int wave = tid >> 5;      // 0..7
  const int wm   = wave & 3;      // 4 waves along M
  const int wn   = wave >> 2;     // 2 waves along N
  const int m0   = blockIdx.y * BM;
  const int n0   = blockIdx.x * BN;
  const int l15  = lane & 15;
  const int hsel = lane >> 4;

  // ---- staging geometry: 16B chunks, 3 async b128 per thread per tile ----
  const int c16   = tid & 3;            // 16B chunk within a 64B row
  const int rowA0 = tid >> 2;           // A rows 0..63
  const int rowA1 = 64 + (tid >> 2);    // A rows 64..127
  const int rowB  = tid >> 2;           // B rows 0..63
  const int gnB   = NGUARD ? min(n0 + rowB, N - 1) : (n0 + rowB);

  const uint64_t baseA = (uint64_t)(uintptr_t)A;
  const uint64_t baseB = (uint64_t)(uintptr_t)B;
  const uint32_t gA0 = ((uint32_t)(m0 + rowA0) * lda) * 2u + c16 * 16u;
  const uint32_t gA1 = ((uint32_t)(m0 + rowA1) * lda) * 2u + c16 * 16u;
  const uint32_t gB  = ((uint32_t)gnB * K) * 2u + c16 * 16u;

  uint32_t ldsA[2], ldsB[2];
  ldsA[0] = (uint32_t)(uintptr_t)&sA[0][0][0];
  ldsA[1] = (uint32_t)(uintptr_t)&sA[1][0][0];
  ldsB[0] = (uint32_t)(uintptr_t)&sB[0][0][0];
  ldsB[1] = (uint32_t)(uintptr_t)&sB[1][0][0];
  const uint32_t dA0 = (uint32_t)rowA0 * PADB + c16 * 16u;
  const uint32_t dA1 = (uint32_t)rowA1 * PADB + c16 * 16u;
  const uint32_t dB  = (uint32_t)rowB  * PADB + c16 * 16u;

  v8f acc[2][2] = {};

  auto stage = [&](int s, int it) {
    const uint32_t ko = (uint32_t)it * (BK * 2u);    // byte k offset
    async_copy_b128(ldsA[s] + dA0, gA0 + ko, baseA);
    async_copy_b128(ldsA[s] + dA1, gA1 + ko, baseA);
    async_copy_b128(ldsB[s] + dB,  gB  + ko, baseB);
  };

  auto wmma_step = [&](int s) {
#pragma unroll
    for (int mt = 0; mt < 2; ++mt) {
      const int ar = wm * 32 + mt * 16 + l15;     // A row (M)
      union { v16h v; v8h h[2]; } ua;
      // 16-bit A layout: lanes 0-15 K=0..7,16..23 ; lanes 16-31 K=8..15,24..31
      ua.h[0] = *(const v8h*)&sA[s][ar][8 * hsel];
      ua.h[1] = *(const v8h*)&sA[s][ar][8 * hsel + 16];
#pragma unroll
      for (int nt = 0; nt < 2; ++nt) {
        const int br = wn * 32 + nt * 16 + l15;   // B row (N)
        union { v16h v; v8h h[2]; } ub;
        // 16-bit B layout: lanes 0-15 K=0..7,16..23 ; lanes 16-31 K=8..15,24..31
        ub.h[0] = *(const v8h*)&sB[s][br][8 * hsel];
        ub.h[1] = *(const v8h*)&sB[s][br][8 * hsel + 16];
        if (NGUARD) {
          if (n0 + br >= N) { v8h z = {}; ub.h[0] = z; ub.h[1] = z; }
        }
        acc[mt][nt] = __builtin_amdgcn_wmma_f32_16x16x32_f16(
            false, ua.v, false, ub.v, (short)0, acc[mt][nt], false, false);
      }
    }
  };

  const int nk = K / BK;
  stage(0, 0);                               // prologue: tile 0 -> buffer 0

  int buf = 0;
  for (int it = 0; it < nk - 1; ++it) {      // steady state: always prefetch
    stage(buf ^ 1, it + 1);
    ASYNC_WAIT(3);                           // current tile has landed
    __syncthreads();
    wmma_step(buf);
    __syncthreads();
    buf ^= 1;
  }
  ASYNC_WAIT(0);                             // last tile
  __syncthreads();
  wmma_step(buf);

  // ---- epilogue + store (C/D layout: lane n=l&15, row = vgpr + 8*hsel) ----
#pragma unroll
  for (int mt = 0; mt < 2; ++mt) {
#pragma unroll
    for (int nt = 0; nt < 2; ++nt) {
      const int col = n0 + wn * 32 + nt * 16 + l15;
      if (!NGUARD || col < N) {
#pragma unroll
        for (int r = 0; r < 8; ++r) {
          const int row = m0 + wm * 32 + mt * 16 + 8 * hsel + r;
          float v = acc[mt][nt][r];
          if (EPI == EPI_SOFTPLUS_BIAS) {
            v += bias[col];
            v = (v > 20.0f) ? v : log1pf(__expf(v));      // softplus
          } else if (EPI == EPI_RESID) {
            v += resid[(size_t)row * ldr + col];
          }
          C[(size_t)row * ldc + col] = v;
          if (W16) C16[(size_t)row * ldc + col] = (_Float16)v;
        }
      }
    }
  }
}

// --------------------------- fp32 -> f16 convert ---------------------------
__global__ __launch_bounds__(256)
void cvt_f16_kernel(const float* __restrict__ src, _Float16* __restrict__ dst,
                    int n) {
  const int i = (blockIdx.x * 256 + threadIdx.x) * 4;
  if (i < n) {
    const float4 f = *(const float4*)&src[i];
    dst[i + 0] = (_Float16)f.x; dst[i + 1] = (_Float16)f.y;
    dst[i + 2] = (_Float16)f.z; dst[i + 3] = (_Float16)f.w;
  }
}

// ----------------------- causal depthwise conv + SiLU ----------------------
__global__ __launch_bounds__(256)
void conv_silu_kernel(const float* __restrict__ xz,       // [L, 2*D_INNER]
                      const float* __restrict__ w,        // [D_INNER, 1, 4]
                      const float* __restrict__ b,        // [D_INNER]
                      float* __restrict__ xc,             // [L, D_INNER] f32
                      _Float16* __restrict__ xc16) {      // [L, D_INNER] f16
  const int idx = blockIdx.x * 256 + threadIdx.x;
  const int d = idx & (D_INNER - 1);
  const int l = idx >> 10;
  float acc = b[d];
#pragma unroll
  for (int j = 0; j < 4; ++j) {
    const int ll = l - 3 + j;
    if (ll >= 0) acc += xz[(size_t)ll * (2 * D_INNER) + d] * w[d * 4 + j];
  }
  const float s = acc / (1.0f + __expf(-acc));            // SiLU
  xc[idx]   = s;
  xc16[idx] = (_Float16)s;
}

// ----------------------------- windowed scan -------------------------------
// block = (64, 4): x = state index n, y picks one of 4 channels d.
// Shift registers of e_j = exp(dt_j * A[d,n]) and dt*B*x give the K=8 window
// via running products: exactly one exp per (t,d,n).
__global__ __launch_bounds__(256)
void scan_kernel(const float* __restrict__ xc,     // [L, D_INNER]
                 const float* __restrict__ dt,     // [L, D_INNER]
                 const float* __restrict__ xdbl,   // [L, 160]: dt_raw|B|C
                 const float* __restrict__ A_log,  // [D_INNER, STATE]
                 const float* __restrict__ Dp,     // [D_INNER]
                 const float* __restrict__ xz,     // [L, 2*D_INNER] (z half)
                 _Float16* __restrict__ y16) {     // [L, D_INNER] f16
  const int n = threadIdx.x;                       // 0..63
  const int d = blockIdx.x * 4 + threadIdx.y;
  const float a  = -__expf(A_log[(size_t)d * STATE + n]);
  const float dD = Dp[d];

  float e[K_WIN] = {}, g[K_WIN] = {};              // shift registers (regs)
  __shared__ float partial[8];

  for (int t = 0; t < SEQ_L; ++t) {
    const float dt_t = dt[(size_t)t * D_INNER + d];
    const float x_t  = xc[(size_t)t * D_INNER + d];
    const float Bn   = xdbl[(size_t)t * XDBL_N + DT_RANK + n];
    const float Cn   = xdbl[(size_t)t * XDBL_N + DT_RANK + STATE + n];

#pragma unroll
    for (int k = K_WIN - 1; k >= 1; --k) { e[k] = e[k - 1]; g[k] = g[k - 1]; }
    e[0] = __expf(dt_t * a);
    g[0] = dt_t * Bn * x_t;

    float h = g[0], p = 1.0f;
#pragma unroll
    for (int k = 1; k < K_WIN; ++k) { p *= e[k - 1]; h += p * g[k]; }

    float v = h * Cn;                                // contribution to <h,C>
#pragma unroll
    for (int off = 16; off > 0; off >>= 1) v += __shfl_down(v, off, 32);
    const int wv = (threadIdx.y * 64 + threadIdx.x) >> 5;  // 0..7
    if ((threadIdx.x & 31) == 0) partial[wv] = v;
    __syncthreads();
    if (threadIdx.x == 0) {
      const float s  = partial[threadIdx.y * 2] + partial[threadIdx.y * 2 + 1];
      const float yv = s + dD * x_t;
      const float zv = xz[(size_t)t * (2 * D_INNER) + D_INNER + d];
      const float sz = zv / (1.0f + __expf(-zv));    // SiLU(z)
      y16[(size_t)t * D_INNER + d] = (_Float16)(yv * sz);
    }
    __syncthreads();
  }
}

// ------------------------------- LayerNorm ---------------------------------
__global__ __launch_bounds__(256)
void layernorm_kernel(const float* __restrict__ in,   // [L, D_MODEL]
                      const float* __restrict__ w,
                      const float* __restrict__ b,
                      float* __restrict__ out) {
  const int row = blockIdx.x;
  const int tid = threadIdx.x;
  const float v0 = in[(size_t)row * D_MODEL + tid];
  const float v1 = in[(size_t)row * D_MODEL + tid + 256];
  float s = v0 + v1, s2 = v0 * v0 + v1 * v1;
  __shared__ float rs[8], rs2[8], stat[2];
#pragma unroll
  for (int off = 16; off > 0; off >>= 1) {
    s  += __shfl_down(s,  off, 32);
    s2 += __shfl_down(s2, off, 32);
  }
  if ((tid & 31) == 0) { rs[tid >> 5] = s; rs2[tid >> 5] = s2; }
  __syncthreads();
  if (tid == 0) {
    float ts = 0.f, ts2 = 0.f;
#pragma unroll
    for (int i = 0; i < 8; ++i) { ts += rs[i]; ts2 += rs2[i]; }
    const float mu  = ts * (1.0f / D_MODEL);
    const float var = ts2 * (1.0f / D_MODEL) - mu * mu;
    stat[0] = mu;
    stat[1] = rsqrtf(var + 1e-5f);
  }
  __syncthreads();
  const float mu = stat[0], inv = stat[1];
  out[(size_t)row * D_MODEL + tid]       = (v0 - mu) * inv * w[tid] + b[tid];
  out[(size_t)row * D_MODEL + tid + 256] = (v1 - mu) * inv * w[tid + 256] + b[tid + 256];
}

// ------------------------------- launcher ----------------------------------
extern "C" void kernel_launch(void* const* d_in, const int* in_sizes, int n_in,
                              void* d_out, int out_size, void* d_ws, size_t ws_size,
                              hipStream_t stream) {
  const float* x         = (const float*)d_in[0];   // [1,1024,512]
  const float* in_proj_w = (const float*)d_in[1];   // [2048, 512]
  const float* conv_w    = (const float*)d_in[2];   // [1024, 1, 4]
  const float* conv_b    = (const float*)d_in[3];   // [1024]
  const float* x_proj_w  = (const float*)d_in[4];   // [160, 1024]
  const float* dt_proj_w = (const float*)d_in[5];   // [1024, 32]
  const float* dt_proj_b = (const float*)d_in[6];   // [1024]
  const float* A_log     = (const float*)d_in[7];   // [1024, 64]
  const float* D_param   = (const float*)d_in[8];   // [1024]
  const float* out_proj_w= (const float*)d_in[9];   // [512, 1024]
  const float* ln_w      = (const float*)d_in[10];  // [512]
  const float* ln_b      = (const float*)d_in[11];  // [512]
  float* out = (float*)d_out;                       // [1,1024,512]

  // ---- workspace layout ----
  float* xz   = (float*)d_ws;                         // [1024, 2048]
  float* xc   = xz   + (size_t)SEQ_L * 2 * D_INNER;   // [1024, 1024]
  float* xdbl = xc   + (size_t)SEQ_L * D_INNER;       // [1024, 160]
  float* dt   = xdbl + (size_t)SEQ_L * XDBL_N;        // [1024, 1024]
  float* pre  = xz;                                   // reuse after scan
  _Float16* hp     = (_Float16*)(dt + (size_t)SEQ_L * D_INNER);
  _Float16* x16    = hp;  hp += (size_t)SEQ_L * D_MODEL;
  _Float16* w_in16 = hp;  hp += (size_t)2 * D_INNER * D_MODEL;
  _Float16* w_xp16 = hp;  hp += (size_t)XDBL_N * D_INNER;
  _Float16* w_dt16 = hp;  hp += (size_t)D_INNER * DT_RANK;
  _Float16* w_out16= hp;  hp += (size_t)D_MODEL * D_INNER;
  _Float16* xc16   = hp;  hp += (size_t)SEQ_L * D_INNER;
  _Float16* xdbl16 = hp;  hp += (size_t)SEQ_L * XDBL_N;
  _Float16* y16    = hp;  hp += (size_t)SEQ_L * D_INNER;

  // ---- 0) one-time fp32 -> f16 conversions ----
  cvt_f16_kernel<<<(SEQ_L * D_MODEL) / 1024, 256, 0, stream>>>(x, x16, SEQ_L * D_MODEL);
  cvt_f16_kernel<<<(2 * D_INNER * D_MODEL) / 1024, 256, 0, stream>>>(in_proj_w, w_in16, 2 * D_INNER * D_MODEL);
  cvt_f16_kernel<<<(XDBL_N * D_INNER) / 1024, 256, 0, stream>>>(x_proj_w, w_xp16, XDBL_N * D_INNER);
  cvt_f16_kernel<<<(D_INNER * DT_RANK) / 1024, 256, 0, stream>>>(dt_proj_w, w_dt16, D_INNER * DT_RANK);
  cvt_f16_kernel<<<(D_MODEL * D_INNER) / 1024, 256, 0, stream>>>(out_proj_w, w_out16, D_MODEL * D_INNER);

  // ---- 1) xz = x @ in_proj_w^T        M=1024 N=2048 K=512 ----
  gemm_nt_wmma<EPI_NONE, false, false><<<dim3(2 * D_INNER / BN, SEQ_L / BM), 256, 0, stream>>>(
      x16, D_MODEL, w_in16, xz, 2 * D_INNER, nullptr,
      SEQ_L, 2 * D_INNER, D_MODEL, nullptr, nullptr, 0);

  // ---- 2) causal depthwise conv + SiLU ----
  conv_silu_kernel<<<(SEQ_L * D_INNER) / 256, 256, 0, stream>>>(
      xz, conv_w, conv_b, xc, xc16);

  // ---- 3) x_dbl = xc @ x_proj_w^T     M=1024 N=160 K=1024 (N guarded) ----
  gemm_nt_wmma<EPI_NONE, true, true><<<dim3((XDBL_N + BN - 1) / BN, SEQ_L / BM), 256, 0, stream>>>(
      xc16, D_INNER, w_xp16, xdbl, XDBL_N, xdbl16,
      SEQ_L, XDBL_N, D_INNER, nullptr, nullptr, 0);

  // ---- 4) dt = softplus(dt_raw @ dt_proj_w^T + b)   M=1024 N=1024 K=32 ----
  gemm_nt_wmma<EPI_SOFTPLUS_BIAS, false, false><<<dim3(D_INNER / BN, SEQ_L / BM), 256, 0, stream>>>(
      xdbl16, XDBL_N, w_dt16, dt, D_INNER, nullptr,
      SEQ_L, D_INNER, DT_RANK, dt_proj_b, nullptr, 0);

  // ---- 5) windowed selective scan, fused * SiLU(z), writes f16 y ----
  scan_kernel<<<D_INNER / 4, dim3(64, 4), 0, stream>>>(
      xc, dt, xdbl, A_log, D_param, xz, y16);

  // ---- 6) pre = y @ out_proj_w^T + residual   M=1024 N=512 K=1024 ----
  gemm_nt_wmma<EPI_RESID, false, false><<<dim3(D_MODEL / BN, SEQ_L / BM), 256, 0, stream>>>(
      y16, D_INNER, w_out16, pre, D_MODEL, nullptr,
      SEQ_L, D_MODEL, D_INNER, nullptr, x, D_MODEL);

  // ---- 7) LayerNorm ----
  layernorm_kernel<<<SEQ_L, 256, 0, stream>>>(pre, ln_w, ln_b, out);
}
